// GP_model_17660905521579
// MI455X (gfx1250) — matrix operand order repaired and checked by reading, not compiled, not run on hardware
//
#include <hip/hip_runtime.h>

// ---------------- problem constants ----------------
#define Bsz   32
#define Tsz   512
#define Hsz   1024
#define H3    3072
#define NTOK  26
#define NOUT  26
#define ROWS  (Bsz * Tsz)      // 16384
#define TK    32               // GEMM k-slab
#define TN    64               // GEMM workgroup N tile

// ---------------- CDNA5 WMMA types ----------------
typedef __bf16 bf16_t;
typedef bf16_t       bf16x16 __attribute__((ext_vector_type(16)));
typedef float        f32x8   __attribute__((ext_vector_type(8)));
typedef unsigned int u32x4   __attribute__((ext_vector_type(4)));

union FragB16 {
  bf16x16 v;
  u32x4   q[2];
};

__device__ __forceinline__ unsigned short f2bf(float f) {
  unsigned int u = __float_as_uint(f);
  u += 0x7fffu + ((u >> 16) & 1u);     // round-to-nearest-even
  return (unsigned short)(u >> 16);
}

__device__ __forceinline__ float sigm(float x) { return 1.0f / (1.0f + expf(-x)); }

// Async DMA one 16-byte chunk per lane: global -> LDS (gfx1250, ASYNCcnt-tracked).
__device__ __forceinline__ void async_ld_b128(unsigned lds_off, unsigned long long gaddr) {
  asm volatile("global_load_async_to_lds_b128 %0, %1, off"
               :: "v"(lds_off), "v"(gaddr) : "memory");
}

__device__ __forceinline__ void wait_async0() {
#if __has_builtin(__builtin_amdgcn_s_wait_asynccnt)
  __builtin_amdgcn_s_wait_asynccnt(0);
#else
  asm volatile("s_wait_asynccnt 0x0" ::: "memory");
#endif
}

// A-matrix 16x32 bf16 fragment (ISA 7.12.2): lane L<16 holds row L, K in
// {hi*8..hi*8+7} u {16+hi*8..16+hi*8+7} with hi = lane>>4.
__device__ __forceinline__ bf16x16 load_a_frag(const unsigned short* X, int ld,
                                               int m0, int k0, int lane) {
  const int l = lane & 15, hi = lane >> 4;
  const unsigned short* p = X + (size_t)(m0 + l) * ld + k0 + hi * 8;
  FragB16 f;
  f.q[0] = *(const u32x4*)(p);        // K = hi*8 .. hi*8+7
  f.q[1] = *(const u32x4*)(p + 16);   // K = 16+hi*8 .. 16+hi*8+7
  return f.v;
}

// B-matrix 32x16 bf16 fragment from global: lane holds column n0+(lane&15), K in
// {hi*8..hi*8+7} u {16+hi*8..16+hi*8+7} with hi = lane>>4.
__device__ __forceinline__ bf16x16 load_b_frag(const unsigned short* Bt, int ld,
                                               int k0, int n0, int lane) {
  const unsigned short* p = Bt + (size_t)(k0 + (lane >> 4) * 8) * ld + n0 + (lane & 15);
  FragB16 f;
  #pragma unroll
  for (int s = 0; s < 16; ++s)
    f.v[s] = __builtin_bit_cast(bf16_t, p[(size_t)(s < 8 ? s : s + 8) * ld]);
  return f.v;
}

__device__ __forceinline__ f32x8 wmma_bf16(bf16x16 a, bf16x16 b, f32x8 c) {
  return __builtin_amdgcn_wmma_f32_16x16x32_bf16(false, a, false, b,
                                                 (short)0, c, false, false);
}

// ---------------- tiny prep kernels ----------------

// Fold SharedLinear(TOK->256)/(DIR->256) into W_emb:  Wt1 = W_emb[:, :256] @ W_tok
__global__ void fusew_k(const float* __restrict__ W_emb, const float* __restrict__ W_tok,
                        const float* __restrict__ W_coord,
                        float* __restrict__ Wt1, float* __restrict__ Wt2) {
  int idx = blockIdx.x * 256 + threadIdx.x;
  if (idx >= Hsz * NTOK) return;
  int h = idx / NTOK, j = idx % NTOK;
  float s1 = 0.f, s2 = 0.f;
  for (int e = 0; e < 256; ++e) {
    s1 += W_emb[(size_t)h * 512 + e]       * W_tok[(size_t)e * NTOK + j];
    s2 += W_emb[(size_t)h * 512 + 256 + e] * W_coord[(size_t)e * NTOK + j];
  }
  Wt1[idx] = s1;
  Wt2[idx] = s2;
}

// emb = relu(t @ Wt1^T + c @ Wt2^T + b_emb)  -> bf16 (WMMA A input), K=26 so VALU is fine
__global__ void emb_k(const float* __restrict__ t_in, const float* __restrict__ c_in,
                      const float* __restrict__ Wt1, const float* __restrict__ Wt2,
                      const float* __restrict__ b_emb, unsigned short* __restrict__ X0) {
  size_t idx = (size_t)blockIdx.x * 256 + threadIdx.x;
  if (idx >= (size_t)ROWS * Hsz) return;
  int h = (int)(idx & (Hsz - 1));
  size_t i = idx >> 10;
  const float* tr = t_in + i * NTOK;
  const float* cr = c_in + i * NTOK;
  const float* w1 = Wt1 + (size_t)h * NTOK;
  const float* w2 = Wt2 + (size_t)h * NTOK;
  float s = b_emb[h];
  #pragma unroll
  for (int j = 0; j < NTOK; ++j) s += tr[j] * w1[j] + cr[j] * w2[j];
  X0[idx] = f2bf(fmaxf(s, 0.f));
}

// Bt[k*Npad + n] = bf16(W[n*K + k])  (weights -> row-major "B layout", zero-padded cols)
__global__ void convbt_k(const float* __restrict__ W, unsigned short* __restrict__ Bt,
                         int N, int K, int Npad) {
  size_t idx = (size_t)blockIdx.x * 256 + threadIdx.x;
  if (idx >= (size_t)K * Npad) return;
  int n = (int)(idx % Npad);
  size_t k = idx / Npad;
  float v = (n < N) ? W[(size_t)n * K + k] : 0.f;
  Bt[idx] = f2bf(v);
}

// ---------------- bulk WMMA GEMM:  C = act(A_bf16 @ Bt_bf16 + bias) ----------------
// WG = 256 threads = 8 waves arranged 4(M) x 2(N); wave tile 32x32; WG tile 128x64.
// B k-slabs are DMA'd global->LDS with async loads (double-buffered, ASYNCcnt),
// shared across all 8 waves; A fragments are loaded directly.
// Requires N % 64 == 0 (pad weights); store mask via nlimit.
__global__ __launch_bounds__(256) void gemm_bt(
    const unsigned short* __restrict__ A,   // (M, K) bf16 row-major
    const unsigned short* __restrict__ Bt,  // (K, N) bf16 row-major
    int M, int N, int K,
    const float* __restrict__ bias, int relu,
    float* __restrict__ Cf, int ldc, int nlimit,
    unsigned short* __restrict__ Cb, int ldcb) {
  __shared__ unsigned short smB[2][TK][TN];  // 2 x 4 KB

  const int tid  = threadIdx.x;
  const int lane = tid & 31;
  const int w    = tid >> 5;
  const int m_w  = blockIdx.y * 128 + (w >> 1) * 32;
  const int n_wg = blockIdx.x * TN;
  const int nsub = (w & 1) * 32;

  // one async b128 per thread covers the whole 32x64 bf16 slab
  const int srow = tid >> 3, schunk = (tid & 7) * 8;

  async_ld_b128((unsigned)(size_t)&smB[0][srow][schunk],
                (unsigned long long)(size_t)(Bt + (size_t)srow * N + n_wg + schunk));
  wait_async0();

  f32x8 acc[2][2] = {};
  int buf = 0;
  for (int k0 = 0; k0 < K; k0 += TK) {
    __syncthreads();  // slab[buf] fully staged; slab[buf^1] no longer being read
    if (k0 + TK < K) {
      async_ld_b128((unsigned)(size_t)&smB[buf ^ 1][srow][schunk],
                    (unsigned long long)(size_t)(Bt + (size_t)(k0 + TK + srow) * N + n_wg + schunk));
      __builtin_prefetch(A + (size_t)(m_w + (lane & 15)) * K + k0 + TK, 0, 1);
    }
    bf16x16 a0 = load_a_frag(A, K, m_w, k0, lane);
    bf16x16 a1 = load_a_frag(A, K, m_w + 16, k0, lane);
    const unsigned short* ls = &smB[buf][(lane >> 4) * 8][nsub + (lane & 15)];  // lane = N-column, K = hi*8.. / 16+hi*8..
    FragB16 fb0, fb1;
    #pragma unroll
    for (int s = 0; s < 16; ++s) {
      const int kk = (s < 8 ? s : s + 8) * TN;
      fb0.v[s] = __builtin_bit_cast(bf16_t, ls[kk]);
      fb1.v[s] = __builtin_bit_cast(bf16_t, ls[kk + 16]);
    }
    acc[0][0] = wmma_bf16(a0, fb0.v, acc[0][0]);
    acc[1][0] = wmma_bf16(a1, fb0.v, acc[1][0]);
    acc[0][1] = wmma_bf16(a0, fb1.v, acc[0][1]);
    acc[1][1] = wmma_bf16(a1, fb1.v, acc[1][1]);
    wait_async0();  // next slab's DMA (this wave's share) complete
    buf ^= 1;
  }

  #pragma unroll
  for (int mi = 0; mi < 2; ++mi) {
    #pragma unroll
    for (int ni = 0; ni < 2; ++ni) {
      const int n0    = n_wg + nsub + ni * 16;
      const int ncol  = n0 + (lane & 15);
      const int mbase = m_w + mi * 16 + (lane >> 4) * 8;
      f32x8 c = acc[mi][ni];
      if (bias) {
        const float bv = bias[ncol];
        #pragma unroll
        for (int i = 0; i < 8; ++i) c[i] += bv;
      }
      if (relu) {
        #pragma unroll
        for (int i = 0; i < 8; ++i) c[i] = fmaxf(c[i], 0.f);
      }
      if (Cf && ncol < nlimit) {
        float* p = Cf + (size_t)mbase * ldc + ncol;
        #pragma unroll
        for (int i = 0; i < 8; ++i) p[(size_t)i * ldc] = c[i];
      }
      if (Cb) {
        unsigned short* p = Cb + (size_t)mbase * ldcb + ncol;
        #pragma unroll
        for (int i = 0; i < 8; ++i) p[(size_t)i * ldcb] = f2bf(c[i]);
      }
    }
  }
}

// ---------------- persistent GRU layer (recurrence in one kernel) ----------------
// 16 WGs x 256 threads. WG owns a 64-wide H tile; wave (m_sub = w>>2, n_sub = w&3)
// computes the r/z/n gate tiles for the SAME (batch-tile, h-tile) so the gate math
// fuses in-register. Hidden state double-buffered in global (L2-resident, 256 KB);
// one atomic grid barrier per timestep.
__global__ __launch_bounds__(256) void gru_layer(
    const float* __restrict__ i_lin,        // (B*T, 3H) precomputed x @ W_ih^T + b_ih
    const unsigned short* __restrict__ WhhT,// (H, 3H) bf16
    const float* __restrict__ b_hh,         // (3H)
    float* __restrict__ acts, float* __restrict__ zg,
    float* __restrict__ rg,   float* __restrict__ ng,  // (B,T,H) slices of d_out
    float* __restrict__ hf,                 // 2 x (32*1024) f32, zero-initialized
    unsigned short* __restrict__ hb,        // 2 x (32*1024) bf16, zero-initialized
    unsigned short* __restrict__ x_next,    // (B*T, H) bf16 for next layer (or null)
    unsigned int* __restrict__ counter) {
  const int lane = threadIdx.x & 31;
  const int w    = threadIdx.x >> 5;
  const int m0   = (w >> 2) * 16;                  // batch tile: 0 or 16
  const int j0   = blockIdx.x * 64 + (w & 3) * 16; // h-column tile
  const int hi   = lane >> 4;
  const int jcol = j0 + (lane & 15);
  const unsigned nblocks = gridDim.x;

  const float bR = b_hh[jcol];
  const float bZ = b_hh[Hsz + jcol];
  const float bN = b_hh[2 * Hsz + jcol];

  for (int t = 0; t < Tsz; ++t) {
    const int cur = t & 1, nxt = cur ^ 1;
    const unsigned short* h_src = hb + (size_t)cur * (Bsz * Hsz);

    f32x8 aR = {}, aZ = {}, aN = {};
    for (int k0 = 0; k0 < Hsz; k0 += 32) {
      bf16x16 a  = load_a_frag(h_src, Hsz, m0, k0, lane);
      bf16x16 br = load_b_frag(WhhT, H3, k0, j0, lane);
      bf16x16 bz = load_b_frag(WhhT, H3, k0, Hsz + j0, lane);
      bf16x16 bn = load_b_frag(WhhT, H3, k0, 2 * Hsz + j0, lane);
      aR = wmma_bf16(a, br, aR);
      aZ = wmma_bf16(a, bz, aZ);
      aN = wmma_bf16(a, bn, aN);
    }

    #pragma unroll
    for (int i = 0; i < 8; ++i) {
      const int b = m0 + hi * 8 + i;
      const size_t row = (size_t)b * Tsz + t;
      const float iR = i_lin[row * H3 + jcol];
      const float iZ = i_lin[row * H3 + Hsz + jcol];
      const float iN = i_lin[row * H3 + 2 * Hsz + jcol];
      const float r  = sigm(iR + aR[i] + bR);
      const float z  = sigm(iZ + aZ[i] + bZ);
      const float nn = tanhf(iN + r * (aN[i] + bN));
      const float hp = hf[(size_t)cur * (Bsz * Hsz) + (size_t)b * Hsz + jcol];
      const float hnew = nn + z * (hp - nn);
      const size_t oidx = (size_t)b * (Tsz * Hsz) + (size_t)t * Hsz + jcol;
      acts[oidx] = hnew;
      zg[oidx]   = z;
      rg[oidx]   = r;
      ng[oidx]   = nn;
      hf[(size_t)nxt * (Bsz * Hsz) + (size_t)b * Hsz + jcol] = hnew;
      hb[(size_t)nxt * (Bsz * Hsz) + (size_t)b * Hsz + jcol] = f2bf(hnew);
      if (x_next) x_next[row * Hsz + jcol] = f2bf(hnew);
    }

    // grid-wide barrier: monotonically increasing counter, one arrive per WG.
    __threadfence();
    __syncthreads();
    if (threadIdx.x == 0) {
      __hip_atomic_fetch_add(counter, 1u, __ATOMIC_RELEASE, __HIP_MEMORY_SCOPE_AGENT);
      const unsigned target = nblocks * (unsigned)(t + 1);
      while (__hip_atomic_load(counter, __ATOMIC_ACQUIRE, __HIP_MEMORY_SCOPE_AGENT) < target)
        __builtin_amdgcn_s_sleep(1);
    }
    __syncthreads();
  }
}

// ---------------- layernorm over H, output bf16 ----------------
__global__ __launch_bounds__(256) void layernorm_k(
    const float* __restrict__ x, const float* __restrict__ gamma,
    const float* __restrict__ beta, unsigned short* __restrict__ out) {
  __shared__ float s1[256];
  __shared__ float s2[256];
  const int tid = threadIdx.x;
  const size_t row = blockIdx.x;
  const float* xr = x + row * Hsz;
  float s = 0.f, ss = 0.f;
  for (int j = tid; j < Hsz; j += 256) {
    float v = xr[j];
    s += v;
    ss += v * v;
  }
  s1[tid] = s;
  s2[tid] = ss;
  __syncthreads();
  for (int o = 128; o > 0; o >>= 1) {
    if (tid < o) {
      s1[tid] += s1[tid + o];
      s2[tid] += s2[tid + o];
    }
    __syncthreads();
  }
  const float mu  = s1[0] * (1.0f / Hsz);
  const float var = s2[0] * (1.0f / Hsz) - mu * mu;
  const float rs  = rsqrtf(var + 1e-5f);
  for (int j = tid; j < Hsz; j += 256)
    out[row * Hsz + j] = f2bf((xr[j] - mu) * rs * gamma[j] + beta[j]);
}

// ---------------- host orchestration ----------------
extern "C" void kernel_launch(void* const* d_in, const int* in_sizes, int n_in,
                              void* d_out, int out_size, void* d_ws, size_t ws_size,
                              hipStream_t stream) {
  (void)in_sizes; (void)n_in; (void)out_size; (void)ws_size;
  const float* t_in    = (const float*)d_in[0];
  const float* c_in    = (const float*)d_in[1];
  const float* W_tok   = (const float*)d_in[2];
  const float* W_coord = (const float*)d_in[3];
  const float* W_emb   = (const float*)d_in[4];
  const float* b_emb   = (const float*)d_in[5];
  const float* W_ih    = (const float*)d_in[6];  // (2, 3H, H)
  const float* W_hh    = (const float*)d_in[7];
  const float* b_ih    = (const float*)d_in[8];  // (2, 3H)
  const float* b_hh    = (const float*)d_in[9];
  const float* gamma   = (const float*)d_in[10];
  const float* beta    = (const float*)d_in[11];
  const float* W_fc    = (const float*)d_in[12]; // (H, H)
  const float* W_ro    = (const float*)d_in[13]; // (OUT, H)

  char* ws = (char*)d_ws;
  size_t off = 0;
  auto take = [&](size_t bytes) -> char* {
    char* p = ws + off;
    off += (bytes + 255) & ~(size_t)255;
    return p;
  };

  // zero-initialized region: barrier counters + hidden-state double buffers
  unsigned int*   counters = (unsigned int*)take(256);
  float*          hf       = (float*)take(sizeof(float) * 2 * 2 * Bsz * Hsz);
  unsigned short* hb       = (unsigned short*)take(sizeof(unsigned short) * 2 * 2 * Bsz * Hsz);
  const size_t zero_bytes = off;

  float*          Wt1    = (float*)take(sizeof(float) * Hsz * NTOK);
  float*          Wt2    = (float*)take(sizeof(float) * Hsz * NTOK);
  unsigned short* WihT   = (unsigned short*)take(sizeof(short) * 2 * Hsz * H3);
  unsigned short* WhhT   = (unsigned short*)take(sizeof(short) * 2 * Hsz * H3);
  unsigned short* WfcT   = (unsigned short*)take(sizeof(short) * Hsz * Hsz);
  unsigned short* WroT   = (unsigned short*)take(sizeof(short) * Hsz * 64);   // padded to 64 cols
  unsigned short* X0     = (unsigned short*)take(sizeof(short) * (size_t)ROWS * Hsz);
  unsigned short* X1     = (unsigned short*)take(sizeof(short) * (size_t)ROWS * Hsz);
  unsigned short* normbf = (unsigned short*)take(sizeof(short) * (size_t)ROWS * Hsz);
  unsigned short* fcbf   = (unsigned short*)take(sizeof(short) * (size_t)ROWS * Hsz);
  float*          i_lin  = (float*)take(sizeof(float) * (size_t)ROWS * H3);

  hipMemsetAsync(d_ws, 0, zero_bytes, stream);

  // d_out layout: output | acts | zg | rg | ng
  float* out  = (float*)d_out;
  const size_t OUT_SZ = (size_t)Bsz * Tsz * NOUT;
  const size_t ACT_T  = (size_t)2 * ROWS * Hsz;   // per returned tensor (L,B,T,H)
  const size_t ACT_L  = (size_t)ROWS * Hsz;       // per layer slice
  float* acts = out + OUT_SZ;
  float* zg   = acts + ACT_T;
  float* rg   = zg + ACT_T;
  float* ng   = rg + ACT_T;

  // 1) fold TOK/DIR projections into W_emb, compute emb as bf16
  fusew_k<<<(Hsz * NTOK + 255) / 256, 256, 0, stream>>>(W_emb, W_tok, W_coord, Wt1, Wt2);
  emb_k<<<(unsigned)(((size_t)ROWS * Hsz + 255) / 256), 256, 0, stream>>>(
      t_in, c_in, Wt1, Wt2, b_emb, X0);

  // 2) weights -> bf16 B-layout (transposed, row-major K x N)
  for (int l = 0; l < 2; ++l) {
    convbt_k<<<(unsigned)(((size_t)Hsz * H3 + 255) / 256), 256, 0, stream>>>(
        W_ih + (size_t)l * H3 * Hsz, WihT + (size_t)l * Hsz * H3, H3, Hsz, H3);
    convbt_k<<<(unsigned)(((size_t)Hsz * H3 + 255) / 256), 256, 0, stream>>>(
        W_hh + (size_t)l * H3 * Hsz, WhhT + (size_t)l * Hsz * H3, H3, Hsz, H3);
  }
  convbt_k<<<(unsigned)(((size_t)Hsz * Hsz + 255) / 256), 256, 0, stream>>>(W_fc, WfcT, Hsz, Hsz, Hsz);
  convbt_k<<<(unsigned)(((size_t)Hsz * 64 + 255) / 256), 256, 0, stream>>>(W_ro, WroT, NOUT, Hsz, 64);

  // 3) per-layer: bulk i_lin GEMM, then persistent recurrent kernel
  for (int l = 0; l < 2; ++l) {
    const unsigned short* Xl = (l == 0) ? X0 : X1;
    gemm_bt<<<dim3(H3 / TN, ROWS / 128), 256, 0, stream>>>(
        Xl, WihT + (size_t)l * Hsz * H3, ROWS, H3, Hsz,
        b_ih + (size_t)l * H3, /*relu=*/0,
        i_lin, H3, H3, /*Cb=*/nullptr, 0);
    gru_layer<<<Hsz / 64, 256, 0, stream>>>(
        i_lin, WhhT + (size_t)l * Hsz * H3, b_hh + (size_t)l * H3,
        acts + (size_t)l * ACT_L, zg + (size_t)l * ACT_L,
        rg + (size_t)l * ACT_L, ng + (size_t)l * ACT_L,
        hf + (size_t)l * 2 * Bsz * Hsz, hb + (size_t)l * 2 * Bsz * Hsz,
        (l == 0) ? X1 : nullptr, counters + l * 16);
  }

  // 4) layernorm(top) -> bf16, fc (relu) -> bf16, readout -> d_out
  layernorm_k<<<ROWS, 256, 0, stream>>>(acts + ACT_L, gamma, beta, normbf);
  gemm_bt<<<dim3(Hsz / TN, ROWS / 128), 256, 0, stream>>>(
      normbf, WfcT, ROWS, Hsz, Hsz, nullptr, /*relu=*/1,
      nullptr, 0, 0, fcbf, Hsz);
  gemm_bt<<<dim3(64 / TN, ROWS / 128), 256, 0, stream>>>(
      fcbf, WroT, ROWS, 64, Hsz, nullptr, /*relu=*/0,
      out, NOUT, NOUT, nullptr, 0);
}
